// ReadingModule_88811333746870
// MI455X (gfx1250) — matrix operand order repaired and checked by reading, not compiled, not run on hardware
//
#include <hip/hip_runtime.h>
#include <hip/hip_bf16.h>
#include <math.h>

typedef _Float16 half_t;
typedef __attribute__((ext_vector_type(16))) _Float16 v16h;
typedef __attribute__((ext_vector_type(8)))  float    v8f;

#define NW    10000
#define INDIM 300
#define S     256
#define NB    16      // batch
#define LL    128     // seq len
#define NLEM  3000
#define D4    1024    // 4*S
#define KD    288     // padded 258
#define KIN   320     // padded 300

#define WMMA(acc, a, b) \
  acc = __builtin_amdgcn_wmma_f32_16x16x32_f16(false, (a), false, (b), (short)0, (acc), false, false)

// A fragment: 16x32 f16 tile; lane L holds row (L&15), K slices per ISA A layout.
static __device__ __forceinline__ v16h load_a_frag(const half_t* aBase, int stride, int kt) {
  int lane = threadIdx.x & 31;
  int row = lane & 15, hh = lane >> 4;
  const half_t* p = aBase + (size_t)row * stride + kt * 32 + hh * 8;
  union { v16h v; uint4 q[2]; } u;
  u.q[0] = *(const uint4*)(p);
  u.q[1] = *(const uint4*)(p + 16);
  return u.v;
}
// B fragment: pre-packed so each lane reads 32 contiguous bytes.
static __device__ __forceinline__ v16h load_b_frag(const half_t* pack, int KT, int nt, int kt) {
  int lane = threadIdx.x & 31;
  const half_t* p = pack + (((size_t)nt * KT + kt) * 32 + lane) * 16;
  union { v16h v; uint4 q[2]; } u;
  u.q[0] = *(const uint4*)(p);
  u.q[1] = *(const uint4*)(p + 8);
  return u.v;
}

// Pack fp32 weight [Ksrc x N] row-major -> f16 WMMA-B tiles [NT][KT][32 lanes][16], zero-pad K.
__global__ void pack_b_kernel(const float* __restrict__ W, half_t* __restrict__ out,
                              int N, int Kpad, int Ksrc) {
  int KT = Kpad >> 5, NT = N >> 4;
  size_t total = (size_t)NT * KT * 512;
  for (size_t idx = (size_t)blockIdx.x * blockDim.x + threadIdx.x; idx < total;
       idx += (size_t)gridDim.x * blockDim.x) {
    int jj   = (int)(idx & 15);
    int lane = (int)((idx >> 4) & 31);
    size_t t = idx >> 9;
    int kt = (int)(t % KT);
    int nt = (int)(t / KT);
    int n = nt * 16 + (lane & 15);
    int k = kt * 32 + (lane >> 4) * 8 + (jj & 7) + (jj >> 3) * 16;
    float v = (k < Ksrc) ? W[(size_t)k * N + n] : 0.f;
    out[idx] = (half_t)v;
  }
}

__global__ void zero_kernel(float4* __restrict__ p, size_t n4) {
  float4 z; z.x = z.y = z.z = z.w = 0.f;
  for (size_t i = (size_t)blockIdx.x * blockDim.x + threadIdx.x; i < n4;
       i += (size_t)gridDim.x * blockDim.x) p[i] = z;
}

// we = relu(WE @ Wp + bp) : [10000,300(->320)] x [320,256]
__global__ void proj_we_kernel(const float* __restrict__ WE, const half_t* __restrict__ wpP,
                               const float* __restrict__ bp, float* __restrict__ we) {
  __shared__ __attribute__((aligned(16))) half_t aT[16 * KIN];
  int tid = threadIdx.x;
  int row0 = blockIdx.x * 16;
  for (int idx = tid; idx < 16 * KIN; idx += 256) {
    int r = idx / KIN, k = idx - r * KIN;
    float v = (k < INDIM) ? WE[(size_t)(row0 + r) * INDIM + k] : 0.f;
    aT[idx] = (half_t)v;
  }
  __syncthreads();
  int wave = tid >> 5, lane = tid & 31;
  const int KT = KIN / 32;
  for (int nn = 0; nn < 2; ++nn) {
    int nt = wave * 2 + nn;
    v8f acc = {};
    for (int kt = 0; kt < KT; ++kt) {
      v16h a = load_a_frag(aT, KIN, kt);
      v16h b = load_b_frag(wpP, KT, nt, kt);
      WMMA(acc, a, b);
    }
    int n = nt * 16 + (lane & 15);
    int mh = (lane >> 4) * 8;
    for (int k = 0; k < 8; ++k) {
      int m = k + mh;
      we[(size_t)(row0 + m) * S + n] = fmaxf(acc[k] + bp[n], 0.f);
    }
  }
}

// ctxt = tile(we, B)
__global__ void tile_ctxt_kernel(const float4* __restrict__ we4, float4* __restrict__ out4) {
  const size_t n4 = (size_t)NW * S / 4;
  for (size_t i = (size_t)blockIdx.x * blockDim.x + threadIdx.x; i < n4;
       i += (size_t)gridDim.x * blockDim.x) {
    float4 v = we4[i];
    for (int b = 0; b < NB; ++b) out4[(size_t)b * n4 + i] = v;
  }
}

// Gather enc_in (f16, K padded to 288) forward and length-reversed.
__global__ void gather_encin_kernel(const float* __restrict__ ctxt, const int* __restrict__ seq,
                                    const int* __restrict__ len, int iter,
                                    half_t* __restrict__ ef, half_t* __restrict__ er) {
  size_t total = (size_t)NB * LL * KD;
  for (size_t idx = (size_t)blockIdx.x * blockDim.x + threadIdx.x; idx < total;
       idx += (size_t)gridDim.x * blockDim.x) {
    int k = (int)(idx % KD);
    int row = (int)(idx / KD);
    int b = row >> 7, l = row & 127;
    float v;
    if (k < S)            v = ctxt[((size_t)(b * NW + seq[row])) * S + k];
    else if (k < S + 2)   v = ((k - S) == iter) ? 1.f : 0.f;
    else                  v = 0.f;
    int ln = len[b];
    int rl = (l < ln) ? (ln - 1 - l) : l;
    half_t h = (half_t)v;
    ef[(size_t)row * KD + k] = h;
    er[((size_t)(b * LL + rl)) * KD + k] = h;
  }
}

// xW = enc_in @ Wi + bias : [2048,288] x [288,1024] -> fp32
__global__ void xw_gemm_kernel(const half_t* __restrict__ A, const half_t* __restrict__ Bp,
                               const float* __restrict__ bias, float* __restrict__ out) {
  int row0 = blockIdx.x * 16;
  int tid = threadIdx.x, wave = tid >> 5, lane = tid & 31;
  const int KT = KD / 32; // 9
  const half_t* aBase = A + (size_t)row0 * KD;
  for (int nn = 0; nn < 8; ++nn) {
    int nt = wave * 8 + nn;
    v8f acc = {};
    for (int kt = 0; kt < KT; ++kt) {
      v16h a = load_a_frag(aBase, KD, kt);
      v16h b = load_b_frag(Bp, KT, nt, kt);
      WMMA(acc, a, b);
    }
    int n = nt * 16 + (lane & 15);
    int mh = (lane >> 4) * 8;
    for (int k = 0; k < 8; ++k) {
      int m = k + mh;
      out[(size_t)(row0 + m) * D4 + n] = acc[k] + bias[n];
    }
  }
}

// Persistent LSTM scan: grid=2 (fwd/bwd), 512 threads; h,c,z in LDS; h@Wh via WMMA each step.
__global__ void lstm_scan_kernel(const float* __restrict__ xwF, const float* __restrict__ xwB,
                                 const half_t* __restrict__ whF, const half_t* __restrict__ whB,
                                 float* __restrict__ hsF, float* __restrict__ hsB) {
  extern __shared__ __attribute__((aligned(16))) unsigned char smem_raw[];
  float*  hS  = (float*)smem_raw;            // 16*256
  float*  cS  = hS + 4096;                   // 16*256
  half_t* h16 = (half_t*)(cS + 4096);        // 16*256
  float*  zS  = (float*)(h16 + 4096);        // 16*1024
  const float*  xw = blockIdx.x ? xwB : xwF;
  const half_t* wh = blockIdx.x ? whB : whF;
  float*        hs = blockIdx.x ? hsB : hsF;
  int tid = threadIdx.x, wave = tid >> 5, lane = tid & 31;
  for (int e = tid; e < 4096; e += 512) { hS[e] = 0.f; cS[e] = 0.f; }
  __syncthreads();
  for (int t = 0; t < LL; ++t) {
    for (int e = tid; e < 4096; e += 512) h16[e] = (half_t)hS[e];
    __syncthreads();
    for (int nn = 0; nn < 4; ++nn) {
      int nt = wave * 4 + nn;
      v8f acc = {};
      for (int kt = 0; kt < 8; ++kt) {
        v16h a = load_a_frag(h16, S, kt);
        v16h b = load_b_frag(wh, 8, nt, kt);
        WMMA(acc, a, b);
      }
      int n = nt * 16 + (lane & 15);
      int mh = (lane >> 4) * 8;
      for (int k = 0; k < 8; ++k) {
        int m = k + mh;
        zS[m * D4 + n] = acc[k] + xw[((size_t)(m * LL + t)) * D4 + n];
      }
    }
    __syncthreads();
    for (int e = tid; e < 4096; e += 512) {
      int m = e >> 8, n = e & 255;
      float zi = zS[m * D4 + n];
      float zf = zS[m * D4 + 256 + n];
      float zg = zS[m * D4 + 512 + n];
      float zo = zS[m * D4 + 768 + n];
      float ig = 1.f / (1.f + __expf(-zi));
      float fg = 1.f / (1.f + __expf(-zf));
      float og = 1.f / (1.f + __expf(-zo));
      float c  = fg * cS[e] + ig * tanhf(zg);
      float h  = og * tanhf(c);
      cS[e] = c; hS[e] = h;
      hs[((size_t)(m * LL + t)) * S + n] = h;
    }
    __syncthreads();
  }
}

// hcat f16 [2048,512] = [hf | hb(un-reversed)]
__global__ void hcat_kernel(const float* __restrict__ hsF, const float* __restrict__ hsB,
                            const int* __restrict__ len, half_t* __restrict__ hcat) {
  size_t total = (size_t)NB * LL * 512;
  for (size_t idx = (size_t)blockIdx.x * blockDim.x + threadIdx.x; idx < total;
       idx += (size_t)gridDim.x * blockDim.x) {
    int n = (int)(idx & 511);
    int row = (int)(idx >> 9);
    int b = row >> 7, l = row & 127;
    float v;
    if (n < S) v = hsF[(size_t)row * S + n];
    else {
      int ln = len[b];
      int rl = (l < ln) ? (ln - 1 - l) : l;
      v = hsB[((size_t)(b * LL + rl)) * S + (n - S)];
    }
    hcat[idx] = (half_t)v;
  }
}

// enc = hcat @ Wenc + benc, masked; scatter relu'd values into Lbuf via atomic f32-as-int max.
__global__ void enc_scatter_kernel(const half_t* __restrict__ A, const half_t* __restrict__ Bp,
                                   const float* __restrict__ benc, const int* __restrict__ seq,
                                   const int* __restrict__ len, const int* __restrict__ u2l,
                                   float* __restrict__ Lbuf) {
  __shared__ int lrowS[16];
  __shared__ int validS[16];
  int tid = threadIdx.x;
  int row0 = blockIdx.x * 16;
  if (tid < 16) {
    int j = row0 + tid;
    int b = j >> 7, l = j & 127;
    validS[tid] = (l < len[b]) ? 1 : 0;
    lrowS[tid]  = b * NLEM + u2l[seq[j]];
  }
  __syncthreads();
  int wave = tid >> 5, lane = tid & 31;
  const half_t* aBase = A + (size_t)row0 * 512;
  const int KT = 16;
  for (int nn = 0; nn < 2; ++nn) {
    int nt = wave * 2 + nn;
    v8f acc = {};
    for (int kt = 0; kt < KT; ++kt) {
      v16h a = load_a_frag(aBase, 512, kt);
      v16h b = load_b_frag(Bp, KT, nt, kt);
      WMMA(acc, a, b);
    }
    int n = nt * 16 + (lane & 15);
    int mh = (lane >> 4) * 8;
    for (int k = 0; k < 8; ++k) {
      int m = k + mh;
      if (validS[m]) {
        float v = fmaxf(acc[k] + benc[n], 0.f);
        atomicMax((int*)(Lbuf + (size_t)lrowS[m] * S + n), __float_as_int(v));
      }
    }
  }
}

// lemproj = Lbuf @ Wg2 : [48000,256] x [256,256]
__global__ void lemproj_kernel(const float* __restrict__ Lbuf, const half_t* __restrict__ Bp,
                               float* __restrict__ out) {
  __shared__ __attribute__((aligned(16))) half_t aT[16 * S];
  int tid = threadIdx.x;
  size_t row0 = (size_t)blockIdx.x * 16;
  for (int idx = tid; idx < 16 * S; idx += 256)
    aT[idx] = (half_t)Lbuf[row0 * S + idx];
  __syncthreads();
  int wave = tid >> 5, lane = tid & 31;
  for (int nn = 0; nn < 2; ++nn) {
    int nt = wave * 2 + nn;
    v8f acc = {};
    for (int kt = 0; kt < 8; ++kt) {
      v16h a = load_a_frag(aT, S, kt);
      v16h b = load_b_frag(Bp, 8, nt, kt);
      WMMA(acc, a, b);
    }
    int n = nt * 16 + (lane & 15);
    int mh = (lane >> 4) * 8;
    for (int k = 0; k < 8; ++k)
      out[(row0 + (size_t)(k + mh)) * S + n] = acc[k];
  }
}

// Fused: gate = sigmoid(ctxt@Wg1 + lemproj[gather] + bgate); ctxt = ctxt*g + (1-g)*new_word
__global__ void gate_update_kernel(float* __restrict__ ctxt, const half_t* __restrict__ Bp,
                                   const float* __restrict__ bgate, const float* __restrict__ lemP,
                                   const float* __restrict__ Lbuf, const int* __restrict__ u2l) {
  __shared__ __attribute__((aligned(16))) half_t aT[16 * S];
  __shared__ int lrowS[16];
  int tid = threadIdx.x;
  size_t row0 = (size_t)blockIdx.x * 16;
  if (tid < 16) {
    size_t j = row0 + tid;
    int b = (int)(j / NW);
    int w = (int)(j - (size_t)b * NW);
    lrowS[tid] = b * NLEM + u2l[w];
  }
  for (int idx = tid; idx < 16 * S; idx += 256)
    aT[idx] = (half_t)ctxt[row0 * S + idx];
  if (blockIdx.x + 1 < gridDim.x)
    __builtin_prefetch(ctxt + (row0 + 16) * S + (tid << 4), 0, 0);
  __syncthreads();
  int wave = tid >> 5, lane = tid & 31;
  for (int nn = 0; nn < 2; ++nn) {
    int nt = wave * 2 + nn;
    v8f acc = {};
    for (int kt = 0; kt < 8; ++kt) {
      v16h a = load_a_frag(aT, S, kt);
      v16h b = load_b_frag(Bp, 8, nt, kt);
      WMMA(acc, a, b);
    }
    int n = nt * 16 + (lane & 15);
    int mh = (lane >> 4) * 8;
    for (int k = 0; k < 8; ++k) {
      int m = k + mh;
      size_t lr = (size_t)lrowS[m] * S + n;
      float x = acc[k] + bgate[n] + lemP[lr];
      float g = 1.f / (1.f + __expf(-x));
      float nw = Lbuf[lr];
      size_t cj = (row0 + m) * S + n;
      ctxt[cj] = ctxt[cj] * g + (1.f - g) * nw;
    }
  }
}

extern "C" void kernel_launch(void* const* d_in, const int* in_sizes, int n_in,
                              void* d_out, int out_size, void* d_ws, size_t ws_size,
                              hipStream_t stream) {
  (void)in_sizes; (void)n_in; (void)out_size; (void)ws_size;
  const float* WE    = (const float*)d_in[0];
  const float* Wp    = (const float*)d_in[1];
  const float* bp    = (const float*)d_in[2];
  const float* Wi_f  = (const float*)d_in[3];
  const float* Wh_f  = (const float*)d_in[4];
  const float* b_f   = (const float*)d_in[5];
  const float* Wi_b  = (const float*)d_in[6];
  const float* Wh_b  = (const float*)d_in[7];
  const float* b_b   = (const float*)d_in[8];
  const float* Wenc  = (const float*)d_in[9];
  const float* benc  = (const float*)d_in[10];
  const float* Wgate = (const float*)d_in[11];
  const float* bgate = (const float*)d_in[12];
  const int* seq0 = (const int*)d_in[13];
  const int* seq1 = (const int*)d_in[14];
  const int* len0 = (const int*)d_in[15];
  const int* len1 = (const int*)d_in[16];
  const int* u2l  = (const int*)d_in[17];
  float* ctxt = (float*)d_out;

  char* ws = (char*)d_ws;
  size_t off = 0;
  auto carve = [&](size_t bytes) {
    char* p = ws + off;
    off = (off + bytes + 255) & ~(size_t)255;
    return p;
  };
  half_t* wpP   = (half_t*)carve((size_t)KIN * S * 2);
  half_t* wifP  = (half_t*)carve((size_t)KD * D4 * 2);
  half_t* wibP  = (half_t*)carve((size_t)KD * D4 * 2);
  half_t* whfP  = (half_t*)carve((size_t)S * D4 * 2);
  half_t* whbP  = (half_t*)carve((size_t)S * D4 * 2);
  half_t* wencP = (half_t*)carve((size_t)512 * S * 2);
  half_t* wg1P  = (half_t*)carve((size_t)S * S * 2);
  half_t* wg2P  = (half_t*)carve((size_t)S * S * 2);
  float*  weF   = (float*)carve((size_t)NW * S * 4);
  half_t* ef    = (half_t*)carve((size_t)NB * LL * KD * 2);
  half_t* er    = (half_t*)carve((size_t)NB * LL * KD * 2);
  float*  xwF   = (float*)carve((size_t)NB * LL * D4 * 4);
  float*  xwB   = (float*)carve((size_t)NB * LL * D4 * 4);
  float*  hsF   = (float*)carve((size_t)NB * LL * S * 4);
  float*  hsB   = (float*)carve((size_t)NB * LL * S * 4);
  half_t* hcat  = (half_t*)carve((size_t)NB * LL * 512 * 2);
  float*  Lbuf  = (float*)carve((size_t)NB * NLEM * S * 4);
  float*  lemP  = (float*)carve((size_t)NB * NLEM * S * 4);

  pack_b_kernel<<<256, 256, 0, stream>>>(Wp,   wpP,  S,  KIN, INDIM);
  pack_b_kernel<<<256, 256, 0, stream>>>(Wi_f, wifP, D4, KD,  S + 2);
  pack_b_kernel<<<256, 256, 0, stream>>>(Wi_b, wibP, D4, KD,  S + 2);
  pack_b_kernel<<<256, 256, 0, stream>>>(Wh_f, whfP, D4, S,   S);
  pack_b_kernel<<<256, 256, 0, stream>>>(Wh_b, whbP, D4, S,   S);
  pack_b_kernel<<<256, 256, 0, stream>>>(Wenc, wencP, S, 512, 512);
  pack_b_kernel<<<256, 256, 0, stream>>>(Wgate,                 wg1P, S, S, S);
  pack_b_kernel<<<256, 256, 0, stream>>>(Wgate + (size_t)S * S, wg2P, S, S, S);

  proj_we_kernel<<<NW / 16, 256, 0, stream>>>(WE, wpP, bp, weF);
  tile_ctxt_kernel<<<1024, 256, 0, stream>>>((const float4*)weF, (float4*)ctxt);

  for (int it = 0; it < 2; ++it) {
    const int* seq = it ? seq1 : seq0;
    const int* len = it ? len1 : len0;
    size_t lb4 = (size_t)NB * NLEM * S / 4;
    zero_kernel<<<2048, 256, 0, stream>>>((float4*)Lbuf, lb4);
    zero_kernel<<<2048, 256, 0, stream>>>((float4*)lemP, lb4);
    size_t gtot = (size_t)NB * LL * KD;
    gather_encin_kernel<<<(unsigned)((gtot + 255) / 256), 256, 0, stream>>>(ctxt, seq, len, it, ef, er);
    xw_gemm_kernel<<<NB * LL / 16, 256, 0, stream>>>(ef, wifP, b_f, xwF);
    xw_gemm_kernel<<<NB * LL / 16, 256, 0, stream>>>(er, wibP, b_b, xwB);
    lstm_scan_kernel<<<2, 512, 106496, stream>>>(xwF, xwB, whfP, whbP, hsF, hsB);
    size_t htot = (size_t)NB * LL * 512;
    hcat_kernel<<<(unsigned)((htot + 255) / 256), 256, 0, stream>>>(hsF, hsB, len, hcat);
    enc_scatter_kernel<<<NB * LL / 16, 256, 0, stream>>>(hcat, wencP, benc, seq, len, u2l, Lbuf);
    lemproj_kernel<<<NB * NLEM / 16, 256, 0, stream>>>(Lbuf, wg2P, lemP);
    gate_update_kernel<<<NB * NW / 16, 256, 0, stream>>>(ctxt, wg1P, bgate, lemP, Lbuf, u2l);
  }
}
